// AttentionWithKVCache_50646254354976
// MI455X (gfx1250) — matrix operand order repaired and checked by reading, not compiled, not run on hardware
//
#include <hip/hip_runtime.h>
#include <hip/hip_bf16.h>

// ---------- CDNA5 WMMA types ----------
typedef __attribute__((ext_vector_type(16))) __bf16 bf16x16;
typedef __attribute__((ext_vector_type(8)))  float  f32x8;

#define WMMA_BF16(A, B, C) \
  __builtin_amdgcn_wmma_f32_16x16x32_bf16(false, (A), false, (B), (short)0, (C), false, false)

// ---------- problem constants ----------
constexpr int B_ = 4, S_ = 2048, D_ = 1024;
constexpr int H_ = 16, KVH_ = 2, HD_ = 64, R_ = H_ / KVH_;
constexpr int QKVN = H_ * HD_ + 2 * KVH_ * HD_;   // 1280 (q | k | v)
constexpr int HW_ = 127;                          // half window (W=255)

union FragA { bf16x16 v; float4 q[2]; };          // 16x32 bf16 A/B fragment (32B)
union Bf16x8 { float4 q; __bf16 e[8]; };

// ---------- gfx1250 async global->LDS copy (ASYNCcnt-tracked) ----------
// Builtin signature (from hipcc diagnostic): first param is
// 'int __attribute__((vector_size(16))) __device__ *'  (v4i in AS1),
// second is the LDS-side v4i pointer (AS3), then imm offset, imm cpol.
#if __has_builtin(__builtin_amdgcn_global_load_async_to_lds_b128)
#define HAS_ASYNC_LDS 1
typedef int v4i_ __attribute__((vector_size(16)));
typedef __attribute__((address_space(1))) v4i_* gv4i_p;
typedef __attribute__((address_space(3))) v4i_* lv4i_p;
#else
#define HAS_ASYNC_LDS 0
#endif

__device__ __forceinline__ void async_copy_b128(void* lds_dst, const void* gsrc) {
#if HAS_ASYNC_LDS
  __builtin_amdgcn_global_load_async_to_lds_b128((gv4i_p)(void*)gsrc, (lv4i_p)lds_dst, 0, 0);
#else
  *(float4*)lds_dst = *(const float4*)gsrc;
#endif
}

__device__ __forceinline__ void wait_async() {
#if HAS_ASYNC_LDS
#if __has_builtin(__builtin_amdgcn_s_wait_asynccnt)
  __builtin_amdgcn_s_wait_asynccnt(0);
#else
  asm volatile("s_wait_asynccnt 0" ::: "memory");
#endif
#endif
}

// =====================================================================
// prep: fp32 -> bf16 casts, weight/bias concatenation
// =====================================================================
__global__ void prep_kernel(const float* __restrict__ x,  const float* __restrict__ wq,
                            const float* __restrict__ wk, const float* __restrict__ wv,
                            const float* __restrict__ wo, const float* __restrict__ bq,
                            const float* __restrict__ bk, const float* __restrict__ bv,
                            __bf16* __restrict__ xb, __bf16* __restrict__ wqkvb,
                            __bf16* __restrict__ wob, float* __restrict__ bcat)
{
  long long i = (long long)blockIdx.x * 256 + threadIdx.x;
  const long long n0 = (long long)B_ * S_ * D_;
  const long long n1 = (long long)D_ * QKVN;
  const long long n2 = (long long)D_ * D_;
  if (i < n0) { xb[i] = (__bf16)x[i]; return; }
  i -= n0;
  if (i < n1) {
    int row = (int)(i / QKVN), col = (int)(i % QKVN);
    float v = (col < H_ * HD_) ? wq[row * (H_ * HD_) + col]
            : (col < H_ * HD_ + KVH_ * HD_) ? wk[row * (KVH_ * HD_) + (col - H_ * HD_)]
            : wv[row * (KVH_ * HD_) + (col - H_ * HD_ - KVH_ * HD_)];
    wqkvb[i] = (__bf16)v; return;
  }
  i -= n1;
  if (i < n2) { wob[i] = (__bf16)wo[i]; return; }
  i -= n2;
  if (i < QKVN) {
    int col = (int)i;
    bcat[col] = (col < H_ * HD_) ? bq[col]
              : (col < H_ * HD_ + KVH_ * HD_) ? bk[col - H_ * HD_]
              : bv[col - H_ * HD_ - KVH_ * HD_];
  }
}

// =====================================================================
// Tiled bf16 GEMM: C[M,N] = A[M,K] @ B[K,N] + bias, fp32 accumulate.
// Block = 256 threads = 8 waves; 128x128 tile; wave computes 64x32.
// A tile staged via GLOBAL_LOAD_ASYNC_TO_LDS_B128 (ASYNCcnt); B tile
// transposed in-flight on the sync path.
// =====================================================================
template <bool OUT_BF16>
__global__ void __launch_bounds__(256)
gemm_kernel(const __bf16* __restrict__ A, const __bf16* __restrict__ Bm,
            const float* __restrict__ bias, void* __restrict__ Cout,
            int M, int N, int K)
{
  __shared__ __align__(16) __bf16 lsA[128][40];   // M x 32 (padded)
  __shared__ __align__(16) __bf16 lsBt[128][40];  // N x 32 (B transposed, padded)

  const int tid  = threadIdx.x;
  const int lane = tid & 31, half = lane >> 4, l16 = lane & 15;
  const int w = tid >> 5, wm = w >> 2, wn = w & 3;
  const int bm = blockIdx.y * 128, bn = blockIdx.x * 128;

  const f32x8 zero = {};
  f32x8 acc[4][2];
  for (int i = 0; i < 4; ++i)
    for (int j = 0; j < 2; ++j) acc[i][j] = zero;

  for (int kt = 0; kt < K; kt += 32) {
    // stage A tile (128x32) asynchronously: contiguous 16B per thread
    #pragma unroll
    for (int e = tid * 8; e < 128 * 32; e += 256 * 8) {
      int r = e >> 5, c = e & 31;
      async_copy_b128(&lsA[r][c], A + (long long)(bm + r) * K + kt + c);
    }
    // stage B tile transposed (32x128 -> [n][k]) on the sync path
    #pragma unroll
    for (int e = tid * 8; e < 32 * 128; e += 256 * 8) {
      int kk = e >> 7, nn = e & 127;
      Bf16x8 vv; vv.q = *(const float4*)(Bm + (long long)(kt + kk) * N + bn + nn);
      #pragma unroll
      for (int u = 0; u < 8; ++u) lsBt[nn + u][kk] = vv.e[u];
    }
    if (kt + 32 < K)  // hint next K-slice of B into cache (global_prefetch_b8)
      __builtin_prefetch(Bm + (long long)(kt + 32 + (tid >> 3)) * N + bn + (tid & 7) * 16, 0, 0);
    wait_async();
    __syncthreads();

    FragA af[4], bfr[2];
    #pragma unroll
    for (int i = 0; i < 4; ++i) {
      int m = wm * 64 + i * 16 + l16;
      af[i].q[0] = *(const float4*)&lsA[m][8 * half];        // K = 8h..8h+7
      af[i].q[1] = *(const float4*)&lsA[m][16 + 8 * half];   // K = 16+8h..23+8h
    }
    #pragma unroll
    for (int j = 0; j < 2; ++j) {
      int n = wn * 32 + j * 16 + l16;
      bfr[j].q[0] = *(const float4*)&lsBt[n][8 * half];      // K = 8h..8h+7
      bfr[j].q[1] = *(const float4*)&lsBt[n][16 + 8 * half]; // K = 16+8h..23+8h
    }
    #pragma unroll
    for (int i = 0; i < 4; ++i)
      #pragma unroll
      for (int j = 0; j < 2; ++j)
        acc[i][j] = WMMA_BF16(af[i].v, bfr[j].v, acc[i][j]);
    __syncthreads();
  }

  // epilogue: C/D layout -> VGPR r holds (M = r + 8*half, N = l16)
  #pragma unroll
  for (int i = 0; i < 4; ++i)
    #pragma unroll
    for (int j = 0; j < 2; ++j) {
      int colg = bn + wn * 32 + j * 16 + l16;
      float bvv = bias[colg];
      #pragma unroll
      for (int rr = 0; rr < 8; ++rr) {
        int rowg = bm + wm * 64 + i * 16 + rr + 8 * half;
        float v = acc[i][j][rr] + bvv;
        if (OUT_BF16) ((__bf16*)Cout)[(long long)rowg * N + colg] = (__bf16)v;
        else          ((float*) Cout)[(long long)rowg * N + colg] = v;
      }
    }
}

// =====================================================================
// RoPE in-place on q,k portions of the bf16 qkv buffer
// =====================================================================
__global__ void rope_kernel(__bf16* __restrict__ qkv, const float* __restrict__ fc)
{
  long long i = (long long)blockIdx.x * 256 + threadIdx.x;
  const long long total = (long long)B_ * S_ * (H_ + KVH_) * (HD_ / 2);
  if (i >= total) return;
  int p = (int)(i % (HD_ / 2)); long long t = i / (HD_ / 2);
  int hh = (int)(t % (H_ + KVH_)); t /= (H_ + KVH_);
  int s = (int)(t % S_); int b = (int)(t / S_);
  float c  = fc[s * HD_ + p * 2];
  float sn = fc[s * HD_ + p * 2 + 1];
  long long base = (long long)(b * S_ + s) * QKVN
                 + (hh < H_ ? hh * HD_ : H_ * HD_ + (hh - H_) * HD_) + 2 * p;
  float te = (float)qkv[base], to = (float)qkv[base + 1];
  qkv[base]     = (__bf16)(te * c  - to * sn);
  qkv[base + 1] = (__bf16)(te * sn + to * c);
}

// =====================================================================
// Sliding-window GQA attention, flash-style online softmax, all-WMMA.
// Block = (b, kv, 16-query tile); 8 waves = 8 q-heads sharing K/V in LDS.
// Window keys for the tile: [t-128, t+160) in 9 chunks of 32.
// Chunks 1..7 are provably inside every row's +-127 window -> no masking
// VALU there; only chunks 0 and 8 run the compare/select chain.
// Zero-padded OOB keys -> score 0 (matches reference zero-pad semantics).
// =====================================================================
__global__ void __launch_bounds__(256)
attn_kernel(const __bf16* __restrict__ qkv, __bf16* __restrict__ obuf)
{
  __shared__ __align__(16) __bf16 lsK[32][72];        // 32 keys x 64 hd (padded)
  __shared__ __align__(16) __bf16 lsVt[64][40];       // 64 hd x 32 keys (padded)
  __shared__ __align__(16) __bf16 lsP[8][16][40];     // per-wave 16x32 P scratch

  const int tid  = threadIdx.x;
  const int lane = tid & 31, half = lane >> 4, l16 = lane & 15;
  const int r = tid >> 5;                              // wave id == GQA head index
  int bid = blockIdx.x;
  const int nst = S_ / 16;
  const int st = bid % nst; bid /= nst;
  const int kv = bid % KVH_; const int b = bid / KVH_;
  const int t = st * 16;
  const int head = kv * R_ + r;

  // Q fragments for this wave's 16 rows (A-layout, two HD K-steps)
  FragA aq[2];
  {
    const __bf16* qrow = qkv + (long long)(b * S_ + t + l16) * QKVN + head * HD_;
    aq[0].q[0] = *(const float4*)(qrow + 8 * half);
    aq[0].q[1] = *(const float4*)(qrow + 16 + 8 * half);
    aq[1].q[0] = *(const float4*)(qrow + 32 + 8 * half);
    aq[1].q[1] = *(const float4*)(qrow + 48 + 8 * half);
  }

  const f32x8 zero = {};
  f32x8 accO[4];
  #pragma unroll
  for (int f = 0; f < 4; ++f) accO[f] = zero;
  float mrow[8], lrow[8];
  #pragma unroll
  for (int j = 0; j < 8; ++j) { mrow[j] = -1e30f; lrow[j] = 0.f; }

  for (int c = 0; c < 9; ++c) {
    const int key_base = t - 128 + 32 * c;
    const bool needMask = (c == 0) || (c == 8);               // window edges only
    const bool inBounds = (key_base >= 0) && (key_base + 31 < S_);

    // ---- stage K (row-major) and V (transposed) for this chunk ----
    {
      const int e = tid * 8;                 // 2048 elems = 256 threads * 8
      const int key = e >> 6, hd = e & 63;
      const int kpos = key_base + key;
      const __bf16* row = qkv + (long long)(b * S_ + kpos) * QKVN + H_ * HD_ + kv * HD_;
      float4 v4;
      if (inBounds) {
        async_copy_b128(&lsK[key][hd], row + hd);             // K: async direct copy
        v4 = *(const float4*)(row + KVH_ * HD_ + hd);
      } else {
        float4 k4 = make_float4(0.f, 0.f, 0.f, 0.f);
        v4 = make_float4(0.f, 0.f, 0.f, 0.f);
        if (kpos >= 0 && kpos < S_) {
          k4 = *(const float4*)(row + hd);
          v4 = *(const float4*)(row + KVH_ * HD_ + hd);
        }
        *(float4*)&lsK[key][hd] = k4;
      }
      Bf16x8 vb; vb.q = v4;                                   // V: sync transpose
      #pragma unroll
      for (int u = 0; u < 8; ++u) lsVt[hd + u][key] = vb.e[u];
    }
    wait_async();
    __syncthreads();

    // ---- scores: Q(16x64) @ K^T -> two 16x16 fp32 tiles ----
    f32x8 sc[2];
    #pragma unroll
    for (int sub = 0; sub < 2; ++sub) {
      const int n = sub * 16 + l16;
      FragA bk0, bk1;
      bk0.q[0] = *(const float4*)&lsK[n][8 * half];
      bk0.q[1] = *(const float4*)&lsK[n][16 + 8 * half];
      bk1.q[0] = *(const float4*)&lsK[n][32 + 8 * half];
      bk1.q[1] = *(const float4*)&lsK[n][48 + 8 * half];
      f32x8 s = zero;
      s = WMMA_BF16(aq[0].v, bk0.v, s);
      s = WMMA_BF16(aq[1].v, bk1.v, s);
      sc[sub] = s;
    }

    // ---- mask (edge chunks only) + online softmax update ----
    #pragma unroll
    for (int j = 0; j < 8; ++j) {
      const int qpos = t + j + 8 * half;
      float s0 = sc[0][j], s1 = sc[1][j];
      if (needMask) {
        const int k0 = key_base + l16, k1 = key_base + 16 + l16;
        if (k0 < qpos - HW_ || k0 > qpos + HW_) s0 = -1e30f;
        if (k1 < qpos - HW_ || k1 > qpos + HW_) s1 = -1e30f;
      }
      float rmax = fmaxf(s0, s1);
      #pragma unroll
      for (int d = 1; d < 16; d <<= 1) rmax = fmaxf(rmax, __shfl_xor(rmax, d, 32));
      const float mnew  = fmaxf(mrow[j], rmax);
      const float alpha = __expf(mrow[j] - mnew);
      const float p0 = __expf(s0 - mnew), p1 = __expf(s1 - mnew);
      float rsum = p0 + p1;
      #pragma unroll
      for (int d = 1; d < 16; d <<= 1) rsum += __shfl_xor(rsum, d, 32);
      lrow[j] = lrow[j] * alpha + rsum;
      mrow[j] = mnew;
      #pragma unroll
      for (int f = 0; f < 4; ++f) accO[f][j] *= alpha;
      lsP[r][j + 8 * half][l16]      = (__bf16)p0;   // C-layout -> LDS (row, col)
      lsP[r][j + 8 * half][16 + l16] = (__bf16)p1;
    }
    // intra-wave LDS RAW: explicit CDNA5 split-counter wait
    asm volatile("s_wait_dscnt 0" ::: "memory");

    // ---- PV: P(16x32) @ V(32x64), A-layout reload of P from LDS ----
    FragA ap;
    ap.q[0] = *(const float4*)&lsP[r][l16][8 * half];
    ap.q[1] = *(const float4*)&lsP[r][l16][16 + 8 * half];
    #pragma unroll
    for (int f = 0; f < 4; ++f) {
      const int n = f * 16 + l16;
      FragA bv;
      bv.q[0] = *(const float4*)&lsVt[n][8 * half];
      bv.q[1] = *(const float4*)&lsVt[n][16 + 8 * half];
      accO[f] = WMMA_BF16(ap.v, bv.v, accO[f]);
    }
    __syncthreads();
  }

  // ---- normalize and store (B,S,H,HD) == row-major (B*S) x 1024 ----
  #pragma unroll
  for (int j = 0; j < 8; ++j) {
    const int row = t + j + 8 * half;
    const float rinv = 1.0f / lrow[j];
    #pragma unroll
    for (int f = 0; f < 4; ++f)
      obuf[((long long)(b * S_ + row) * H_ + head) * HD_ + f * 16 + l16] =
          (__bf16)(accO[f][j] * rinv);
  }
}

// =====================================================================
// launch
// =====================================================================
extern "C" void kernel_launch(void* const* d_in, const int* in_sizes, int n_in,
                              void* d_out, int out_size, void* d_ws, size_t ws_size,
                              hipStream_t stream)
{
  const float* x  = (const float*)d_in[0];
  const float* fc = (const float*)d_in[1];
  const float* wq = (const float*)d_in[2];
  const float* bq = (const float*)d_in[3];
  const float* wk = (const float*)d_in[4];
  const float* bk = (const float*)d_in[5];
  const float* wv = (const float*)d_in[6];
  const float* bv = (const float*)d_in[7];
  const float* wo = (const float*)d_in[8];
  const float* bo = (const float*)d_in[9];

  char* ws = (char*)d_ws;
  auto al = [](size_t v) { return (v + 255) & ~(size_t)255; };
  size_t off = 0;
  __bf16* xb    = (__bf16*)(ws + off); off = al(off + sizeof(__bf16) * (size_t)B_ * S_ * D_);
  __bf16* wqkvb = (__bf16*)(ws + off); off = al(off + sizeof(__bf16) * (size_t)D_ * QKVN);
  __bf16* wob   = (__bf16*)(ws + off); off = al(off + sizeof(__bf16) * (size_t)D_ * D_);
  float*  bcat  = (float*)(ws + off);  off = al(off + sizeof(float) * QKVN);
  __bf16* qkv   = (__bf16*)(ws + off); off = al(off + sizeof(__bf16) * (size_t)B_ * S_ * QKVN);
  __bf16* obuf  = (__bf16*)(ws + off); off = al(off + sizeof(__bf16) * (size_t)B_ * S_ * D_);

  const long long totPrep = (long long)B_ * S_ * D_ + (long long)D_ * QKVN
                          + (long long)D_ * D_ + QKVN;
  prep_kernel<<<(int)((totPrep + 255) / 256), 256, 0, stream>>>(
      x, wq, wk, wv, wo, bq, bk, bv, xb, wqkvb, wob, bcat);

  gemm_kernel<true><<<dim3(QKVN / 128, (B_ * S_) / 128), 256, 0, stream>>>(
      xb, wqkvb, bcat, (void*)qkv, B_ * S_, QKVN, D_);

  const long long totRope = (long long)B_ * S_ * (H_ + KVH_) * (HD_ / 2);
  rope_kernel<<<(int)((totRope + 255) / 256), 256, 0, stream>>>(qkv, fc);

  attn_kernel<<<B_ * KVH_ * (S_ / 16), 256, 0, stream>>>(qkv, obuf);

  gemm_kernel<false><<<dim3(D_ / 128, (B_ * S_) / 128), 256, 0, stream>>>(
      obuf, wob, bo, d_out, B_ * S_, D_, D_);
}
